// FFNNTransducer_57526791962913
// MI455X (gfx1250) — matrix operand order repaired and checked by reading, not compiled, not run on hardware
//
#include <hip/hip_runtime.h>
#include <hip/hip_bf16.h>
#include <math.h>
#include <stdint.h>

#define B_   16
#define T_   2000
#define DENC 512
#define V_   1024
#define E_   256
#define P_   640
#define BT   (B_ * T_)
#define ROWS 32                 // rows per workgroup (2 x 16-row M tiles)

typedef __bf16 bf16_t;
typedef __attribute__((ext_vector_type(16))) __bf16 v16bf;
typedef __attribute__((ext_vector_type(8)))  float  v8f;
typedef __attribute__((ext_vector_type(4)))  unsigned int u32x4;

union V16U { v16bf v; u32x4 u[2]; };

#if __has_builtin(__builtin_amdgcn_tanhf)
__device__ __forceinline__ float fast_tanh(float x) { return __builtin_amdgcn_tanhf(x); }
#else
__device__ __forceinline__ float fast_tanh(float x) { return tanhf(x); }
#endif

// ---------------------------------------------------------------------------
// GEMM core with 2x M register blocking:
//   C[32 x N] = A[32 x KDIM] (bf16, LDS, row-major) x Wt[N x KDIM]^T + bias
// Each B fragment (global) is reused for both 16-row M tiles -> halves L2
// weight traffic per row vs. 16-row tiles.
// A frag (16-bit 16x32): lane<16 holds K=[k0..k0+7],[k0+16..k0+23];
//                        lane>=16 holds K=[k0+8..+15],[k0+24..+31]
// B frag (16-bit 32x16): lane n<16 col n K=[k0..k0+7],[k0+16..k0+23]; lane n+16 K=[k0+8..+15],[k0+24..+31]
// ---------------------------------------------------------------------------
template <int KDIM, int NPW>
__device__ __forceinline__ void gemm_acc2(const bf16_t* __restrict__ Alds,
                                          const bf16_t* __restrict__ Wt,
                                          const float*  __restrict__ bias,
                                          int wave, int lane,
                                          v8f acc0[NPW], v8f acc1[NPW]) {
    const int nlo = lane & 15;
    const int sel = lane >> 4;
#pragma unroll
    for (int j = 0; j < NPW; ++j) {
        const float bv = bias[(wave * NPW + j) * 16 + nlo];
#pragma unroll
        for (int i = 0; i < 8; ++i) { acc0[j][i] = bv; acc1[j][i] = bv; }
    }
    const bf16_t* arow0 = Alds + nlo * KDIM + sel * 8;          // rows 0..15
    const bf16_t* arow1 = arow0 + 16 * KDIM;                    // rows 16..31
    for (int k0 = 0; k0 < KDIM; k0 += 32) {
        V16U a0, a1;
        a0.u[0] = *(const u32x4*)(arow0 + k0);
        a0.u[1] = *(const u32x4*)(arow0 + k0 + 16);
        a1.u[0] = *(const u32x4*)(arow1 + k0);
        a1.u[1] = *(const u32x4*)(arow1 + k0 + 16);
#pragma unroll
        for (int j = 0; j < NPW; ++j) {
            const bf16_t* bp = Wt + (size_t)((wave * NPW + j) * 16 + nlo) * KDIM
                                  + k0 + sel * 8;
            V16U b;
            b.u[0] = *(const u32x4*)bp;
            b.u[1] = *(const u32x4*)(bp + 16);
            acc0[j] = __builtin_amdgcn_wmma_f32_16x16x32_bf16(
                false, a0.v, false, b.v, (short)0, acc0[j], false, false);
            acc1[j] = __builtin_amdgcn_wmma_f32_16x16x32_bf16(
                false, a1.v, false, b.v, (short)0, acc1[j], false, false);
        }
    }
}

// C/D layout: VGPR i -> row (i + sel*8), col = tile*16 + (lane&15)
template <int NPW>
__device__ __forceinline__ void store_f32(float* __restrict__ buf, int ldn,
                                          v8f acc0[NPW], v8f acc1[NPW],
                                          int wave, int lane) {
    const int nlo = lane & 15, sel = lane >> 4;
#pragma unroll
    for (int j = 0; j < NPW; ++j) {
        const int n = (wave * NPW + j) * 16 + nlo;
#pragma unroll
        for (int i = 0; i < 8; ++i) {
            buf[(i + sel * 8) * ldn + n]        = acc0[j][i];
            buf[(i + sel * 8 + 16) * ldn + n]   = acc1[j][i];
        }
    }
}

template <int NPW>
__device__ __forceinline__ void store_tanh_bf16(bf16_t* __restrict__ buf, int ldn,
                                                v8f acc0[NPW], v8f acc1[NPW],
                                                int wave, int lane) {
    const int nlo = lane & 15, sel = lane >> 4;
#pragma unroll
    for (int j = 0; j < NPW; ++j) {
        const int n = (wave * NPW + j) * 16 + nlo;
#pragma unroll
        for (int i = 0; i < 8; ++i) {
            buf[(i + sel * 8) * ldn + n]      = (bf16_t)fast_tanh(acc0[j][i]);
            buf[(i + sel * 8 + 16) * ldn + n] = (bf16_t)fast_tanh(acc1[j][i]);
        }
    }
}

// log-softmax over ROWS x 1024 from LDS f32 buffer; each wave reduces 4 rows
// with wave32 shuffles; optional raw bf16 copy of the buffer.
__device__ __forceinline__ void softmax_write(const float* __restrict__ buf,
                                              float* __restrict__ gout,
                                              bf16_t* __restrict__ bcopy,
                                              float* red_mx, float* red_ls,
                                              int tid) {
    const int wave = tid >> 5, lane = tid & 31;
#pragma unroll
    for (int rr = 0; rr < ROWS / 8; ++rr) {
        const int r = wave * (ROWS / 8) + rr;
        float mx = -3.0e38f;
        for (int c = lane; c < V_; c += 32) mx = fmaxf(mx, buf[r * V_ + c]);
#pragma unroll
        for (int off = 16; off > 0; off >>= 1)
            mx = fmaxf(mx, __shfl_xor(mx, off, 32));
        float s = 0.f;
        for (int c = lane; c < V_; c += 32) s += __expf(buf[r * V_ + c] - mx);
#pragma unroll
        for (int off = 16; off > 0; off >>= 1) s += __shfl_xor(s, off, 32);
        if (lane == 0) { red_mx[r] = mx; red_ls[r] = __logf(s); }
    }
    __syncthreads();
    for (int idx = tid; idx < ROWS * V_; idx += 256) {
        const int r = idx >> 10;
        const float v = buf[idx];
        gout[idx] = v - red_mx[r] - red_ls[r];
        if (bcopy) bcopy[idx] = (bf16_t)v;
    }
    __syncthreads();
}

// ---------------------------------------------------------------------------
// Kernel 0: per-batch sequential viterbi-context scan + lens passthrough
// ---------------------------------------------------------------------------
__global__ void ctx_kernel(const int* __restrict__ targets,
                           const int* __restrict__ flen,
                           int* __restrict__ c0, int* __restrict__ c1,
                           int* __restrict__ lens_out) {
    const int b = blockIdx.x;
    if (threadIdx.x != 0) return;
    int cur = 0, prev = 0;
    const int* tg = targets + (size_t)b * T_;
    int* p0 = c0 + (size_t)b * T_;
    int* p1 = c1 + (size_t)b * T_;
    for (int t = 0; t < T_; ++t) {
        p0[t] = cur;
        p1[t] = prev;
        const int nt = tg[t];
        prev = cur;
        if (nt != 0) cur = nt;
    }
    lens_out[b] = flen[b];
}

// ---------------------------------------------------------------------------
// Kernel 1: f32 [K][N] row-major -> bf16 [N][K] row-major (transposed)
// ---------------------------------------------------------------------------
__global__ void transpose_to_bf16(const float* __restrict__ src,
                                  bf16_t* __restrict__ dst, int K, int N) {
    const int idx = blockIdx.x * 256 + threadIdx.x;
    if (idx < K * N) {
        const int k = idx / N, n = idx - k * N;
        dst[(size_t)n * K + k] = (bf16_t)src[idx];
    }
}

// ---------------------------------------------------------------------------
// Kernel 2: fused transducer chain, 32 rows per workgroup, 8 waves
// ---------------------------------------------------------------------------
extern "C" __global__ void __launch_bounds__(256, 1)
fused_transducer(const float* __restrict__ enc_in,
                 const int* __restrict__ ctx0, const int* __restrict__ ctx1,
                 const float* __restrict__ emb,
                 const bf16_t* __restrict__ WencT, const float* __restrict__ b_enc,
                 const bf16_t* __restrict__ Wp1T,  const float* __restrict__ b_p1,
                 const bf16_t* __restrict__ Wp2T,  const float* __restrict__ b_p2,
                 const bf16_t* __restrict__ WpoT,  const float* __restrict__ b_po,
                 const bf16_t* __restrict__ Wj1T,  const float* __restrict__ b_j1,
                 const bf16_t* __restrict__ Wj2T,  const float* __restrict__ b_j2,
                 float* __restrict__ out_log, float* __restrict__ out_pred,
                 float* __restrict__ out_enc) {
    __shared__ bf16_t lds_x [ROWS * 512];    //  32 KB : Xe then Xp (bf16 A tiles)
    __shared__ float  lds_f [ROWS * 1024];   // 128 KB : enc -> pred -> logits (f32)
    __shared__ bf16_t lds_sb[ROWS * 1024];   //  64 KB : enc bf16 copy, then s
    __shared__ bf16_t lds_a1[ROWS * 640];    //  40 KB : tanh act (p1, then joint h)
    __shared__ bf16_t lds_a2[ROWS * 640];    //  40 KB : tanh act (p2)
    __shared__ float  red_mx[ROWS], red_ls[ROWS];

    const int tid  = threadIdx.x;
    const int wave = tid >> 5, lane = tid & 31;
    const size_t row0 = (size_t)blockIdx.x * ROWS;

    // ---- load encoder rows [ROWS x 512] f32 -> bf16 LDS
    {
        const float4* src = (const float4*)(enc_in + row0 * DENC);
        for (int i = tid; i < ROWS * DENC / 4; i += 256) {
            const float4 v = src[i];
            const int o = i * 4;
            lds_x[o]     = (bf16_t)v.x; lds_x[o + 1] = (bf16_t)v.y;
            lds_x[o + 2] = (bf16_t)v.z; lds_x[o + 3] = (bf16_t)v.w;
        }
    }
    __syncthreads();

    // ---- enc = Xe @ W_enc^T + b_enc  (K=512, N=1024)
    { v8f a0[8], a1[8];
      gemm_acc2<512, 8>(lds_x, WencT, b_enc, wave, lane, a0, a1);
      store_f32<8>(lds_f, 1024, a0, a1, wave, lane); }
    __syncthreads();
    softmax_write(lds_f, out_enc + row0 * V_, lds_sb, red_mx, red_ls, tid);

    // ---- gather predictor input: [emb[c0] | emb[c1]] -> bf16 LDS [ROWS x 512]
    {
        for (int i = tid; i < ROWS * DENC / 4; i += 256) {
            const int r = i >> 7, c4 = i & 127;   // 128 float4 per row
            const size_t ridx = row0 + r;
            const int cid = (c4 < 64) ? ctx0[ridx] : ctx1[ridx];
            const float4 v = ((const float4*)(emb + (size_t)cid * E_))[c4 & 63];
            const int o = i * 4;
            lds_x[o]     = (bf16_t)v.x; lds_x[o + 1] = (bf16_t)v.y;
            lds_x[o + 2] = (bf16_t)v.z; lds_x[o + 3] = (bf16_t)v.w;
        }
    }
    __syncthreads();

    // ---- a1 = tanh(Xp @ W_p1^T + b_p1)  (K=512, N=640)
    { v8f a0[5], a1[5];
      gemm_acc2<512, 5>(lds_x, Wp1T, b_p1, wave, lane, a0, a1);
      store_tanh_bf16<5>(lds_a1, 640, a0, a1, wave, lane); }
    __syncthreads();

    // ---- a2 = tanh(a1 @ W_p2^T + b_p2)  (K=640, N=640)
    { v8f a0[5], a1[5];
      gemm_acc2<640, 5>(lds_a1, Wp2T, b_p2, wave, lane, a0, a1);
      store_tanh_bf16<5>(lds_a2, 640, a0, a1, wave, lane); }
    __syncthreads();

    // ---- pred = a2 @ W_po^T + b_po  (K=640, N=1024)
    { v8f a0[8], a1[8];
      gemm_acc2<640, 8>(lds_a2, WpoT, b_po, wave, lane, a0, a1);
      store_f32<8>(lds_f, 1024, a0, a1, wave, lane); }
    __syncthreads();
    softmax_write(lds_f, out_pred + row0 * V_, nullptr, red_mx, red_ls, tid);

    // ---- s = enc + pred (bf16, in place over the enc bf16 copy)
    for (int i = tid; i < ROWS * 1024; i += 256)
        lds_sb[i] = (bf16_t)((float)lds_sb[i] + lds_f[i]);
    __syncthreads();

    // ---- h = tanh(s @ W_j1^T + b_j1)  (K=1024, N=640)
    { v8f a0[5], a1[5];
      gemm_acc2<1024, 5>(lds_sb, Wj1T, b_j1, wave, lane, a0, a1);
      store_tanh_bf16<5>(lds_a1, 640, a0, a1, wave, lane); }
    __syncthreads();

    // ---- logits = h @ W_j2^T + b_j2  (K=640, N=1024)
    { v8f a0[8], a1[8];
      gemm_acc2<640, 8>(lds_a1, Wj2T, b_j2, wave, lane, a0, a1);
      store_f32<8>(lds_f, 1024, a0, a1, wave, lane); }
    __syncthreads();
    softmax_write(lds_f, out_log + row0 * V_, nullptr, red_mx, red_ls, tid);
}

// ---------------------------------------------------------------------------
extern "C" void kernel_launch(void* const* d_in, const int* in_sizes, int n_in,
                              void* d_out, int out_size, void* d_ws, size_t ws_size,
                              hipStream_t stream) {
    const float* enc_in  = (const float*)d_in[0];
    const int*   flen    = (const int*)d_in[1];
    const int*   targets = (const int*)d_in[2];
    const float* W_enc = (const float*)d_in[3];  const float* b_enc = (const float*)d_in[4];
    const float* emb   = (const float*)d_in[5];
    const float* W_p1  = (const float*)d_in[6];  const float* b_p1  = (const float*)d_in[7];
    const float* W_p2  = (const float*)d_in[8];  const float* b_p2  = (const float*)d_in[9];
    const float* W_po  = (const float*)d_in[10]; const float* b_po  = (const float*)d_in[11];
    const float* W_j1  = (const float*)d_in[12]; const float* b_j1  = (const float*)d_in[13];
    const float* W_j2  = (const float*)d_in[14]; const float* b_j2  = (const float*)d_in[15];

    float* out      = (float*)d_out;
    float* out_log  = out;
    float* out_pred = out + (size_t)BT * V_;
    float* out_enc  = out + 2 * (size_t)BT * V_;
    int*   out_lens = (int*)(out + 3 * (size_t)BT * V_);

    // workspace carve-out (~7 MB)
    char* ws = (char*)d_ws;
    int* ctx0 = (int*)ws; ws += sizeof(int) * (size_t)BT;
    int* ctx1 = (int*)ws; ws += sizeof(int) * (size_t)BT;
    ws = (char*)(((uintptr_t)ws + 255) & ~(uintptr_t)255);
    bf16_t* WencT = (bf16_t*)ws; ws += 2 * (size_t)V_ * DENC;
    bf16_t* Wp1T  = (bf16_t*)ws; ws += 2 * (size_t)P_ * DENC;
    bf16_t* Wp2T  = (bf16_t*)ws; ws += 2 * (size_t)P_ * P_;
    bf16_t* WpoT  = (bf16_t*)ws; ws += 2 * (size_t)V_ * P_;
    bf16_t* Wj1T  = (bf16_t*)ws; ws += 2 * (size_t)P_ * V_;
    bf16_t* Wj2T  = (bf16_t*)ws; ws += 2 * (size_t)V_ * P_;

    ctx_kernel<<<B_, 32, 0, stream>>>(targets, flen, ctx0, ctx1, out_lens);

    auto cv = [&](const float* s, bf16_t* d, int K, int N) {
        const int n = K * N;
        transpose_to_bf16<<<(n + 255) / 256, 256, 0, stream>>>(s, d, K, N);
    };
    cv(W_enc, WencT, DENC, V_);
    cv(W_p1,  Wp1T,  DENC, P_);
    cv(W_p2,  Wp2T,  P_,   P_);
    cv(W_po,  WpoT,  P_,   V_);
    cv(W_j1,  Wj1T,  V_,   P_);
    cv(W_j2,  Wj2T,  P_,   V_);

    fused_transducer<<<BT / ROWS, 256, 0, stream>>>(
        enc_in, ctx0, ctx1, emb,
        WencT, b_enc, Wp1T, b_p1, Wp2T, b_p2, WpoT, b_po,
        Wj1T, b_j1, Wj2T, b_j2,
        out_log, out_pred, out_enc);
}
